// MessagePassing_25786983645318
// MI455X (gfx1250) — matrix operand order repaired and checked by reading, not compiled, not run on hardware
//
#include <hip/hip_runtime.h>

// CDNA5 / gfx1250 fused message-passing kernel.
// One workgroup per batch element n (8 WGs), 256 threads = 8 wave32 waves.
// Uses v_wmma_f32_16x16x32_f16 for both GEMM stages; everything LDS-resident
// (306 KB of the 320 KB CDNA5 WGP LDS).

typedef __attribute__((ext_vector_type(16))) _Float16 v16h;
typedef __attribute__((ext_vector_type(8)))  float    v8f;

#define NB   8
#define MA   128
#define HID  32
#define NF   16
#define NCHUNK 17            // chunks 0..15 = mpart (Wc rows 32+32f), chunk 16 = hp (Wc rows 0..31)
#define EPSV 1e-12f
#define UPDR 0.1f

#define SH_B_FLOATS (NCHUNK * MA * HID)          // 69632
#define SH_H_OFF    (SH_B_FLOATS)                // + 4096
#define SH_M_OFF    (SH_H_OFF + MA * HID)        // + 4096
#define SH_R_OFF    (SH_M_OFF + MA * HID)        // + 384
#define SH_MASK_OFF (SH_R_OFF + MA * 3)          // + 128
#define SMEM_FLOATS (SH_MASK_OFF + MA)           // 78336 floats = 313344 B <= 320KB

__global__ void __launch_bounds__(256)
MessagePassing_25786983645318_kernel(const int* __restrict__ z,
                                     const float* __restrict__ r,
                                     const float* __restrict__ h,
                                     const float* __restrict__ dist,
                                     const float* __restrict__ wid,
                                     const float* __restrict__ Wc,
                                     const float* __restrict__ bc,
                                     const float* __restrict__ Wu,
                                     const float* __restrict__ bu,
                                     float* __restrict__ out)
{
    extern __shared__ float smem[];
    float* sB    = smem;                 // [17][128][32] precomputed h @ Wc chunks
    float* sH    = smem + SH_H_OFF;      // [128][32]
    float* sM    = smem + SH_M_OFF;      // [128][32] masked-summed messages
    float* sR    = smem + SH_R_OFF;      // [128][3]
    float* sMask = smem + SH_MASK_OFF;   // [128]

    const int n   = blockIdx.x;
    const int tid = threadIdx.x;
    const int lane = tid & 31;
    const int wv   = tid >> 5;           // wave id 0..7
    const int m16  = lane & 15;
    const int hi   = lane >> 4;          // 0 = lanes 0-15, 1 = lanes 16-31

    int*   outZ = (int*)out;                       // [0, 1024)   : z passthrough (bit copy)
    float* outR = out + NB * MA;                   // [1024, 4096): r passthrough
    float* outH = out + NB * MA + NB * MA * 3;     // [4096, ...) : h_new

    // ---- load inputs to LDS + passthrough copies of z, r -------------------
    {
        const float* hsrc = h + n * MA * HID;
        for (int e = tid; e < MA * HID; e += 256) sH[e] = hsrc[e];
        const float* rsrc = r + n * MA * 3;
        for (int e = tid; e < MA * 3; e += 256) { sR[e] = rsrc[e]; outR[n * MA * 3 + e] = rsrc[e]; }
        const int* zsrc = z + n * MA;
        for (int e = tid; e < MA; e += 256) {
            sMask[e] = (zsrc[e] > -1) ? 1.0f : 0.0f;
            outZ[n * MA + e] = zsrc[e];
        }
    }
    __syncthreads();

    // ---- Stage A: B[cf] = h @ Wc_chunk via WMMA ----------------------------
    // 17 chunks x (8 j-tiles x 2 c-tiles) = 272 16x16 tiles, K=32 -> 1 wmma each.
    for (int t = wv; t < NCHUNK * 16; t += 8) {
        const int cf  = t >> 4;
        const int rem = t & 15;
        const int jt  = rem >> 1;
        const int ct  = rem & 1;
        const int rowbase = (cf == 16) ? 0 : (32 + cf * HID);

        // A fragment: rows M = jt*16 + m16 of h, K interleave per ISA A-layout
        v16h a;
        const float* hr = sH + (jt * 16 + m16) * HID;
#pragma unroll
        for (int u = 0; u < 8; ++u) {
            a[u]     = (_Float16)hr[8 * hi + u];
            a[u + 8] = (_Float16)hr[16 + 8 * hi + u];
        }
        // B fragment: Wc chunk rows (K), column N = ct*16 + m16; K interleave matching A-layout
        v16h b;
        const int col = ct * 16 + m16;
#pragma unroll
        for (int u = 0; u < 8; ++u) {
            b[u]     = (_Float16)Wc[(rowbase + 8 * hi + u) * HID + col];
            b[u + 8] = (_Float16)Wc[(rowbase + 16 + 8 * hi + u) * HID + col];
        }

        v8f c = {};
        c = __builtin_amdgcn_wmma_f32_16x16x32_f16(false, a, false, b, (short)0, c, false, false);

        float* dbase = sB + cf * MA * HID + (jt * 16) * HID;
#pragma unroll
        for (int u = 0; u < 8; ++u)
            dbase[(8 * hi + u) * HID + col] = c[u];
    }
    __syncthreads();

    // ---- Stage B: Gaussian features + SiLU + masked sum over j (VALU) ------
    // thread -> (i = tid>>1, 16-channel half)
    {
        const int i  = tid >> 1;
        const int cb = (tid & 1) * 16;

        float mu[NF], sg[NF];
#pragma unroll
        for (int f = 0; f < NF; ++f) { mu[f] = dist[f]; sg[f] = wid[f]; }

        float a0[16];
#pragma unroll
        for (int c = 0; c < 16; ++c)
            a0[c] = sB[(16 * MA + i) * HID + cb + c] + bc[cb + c];   // h_i @ Wc[0:32] + bc

        float acc[16];
#pragma unroll
        for (int c = 0; c < 16; ++c) acc[c] = 0.0f;

        const float mi = sMask[i];
        const float rx = sR[i * 3 + 0], ry = sR[i * 3 + 1], rz = sR[i * 3 + 2];

        if (mi != 0.0f) {
            for (int j = 0; j < MA; ++j) {
                if (sMask[j] == 0.0f) continue;            // pmask == 0 -> contributes 0
                const float dx = rx - sR[j * 3 + 0];
                const float dy = ry - sR[j * 3 + 1];
                const float dz = rz - sR[j * 3 + 2];
                const float d  = sqrtf(dx * dx + dy * dy + dz * dz + EPSV);
                float g[NF];
#pragma unroll
                for (int f = 0; f < NF; ++f) {
                    const float tdiff = d - mu[f];
                    g[f] = 5.0f * __expf(-(tdiff * tdiff) / sg[f]);
                }
                const float* bj = sB + j * HID + cb;       // chunk stride MA*HID
#pragma unroll
                for (int c = 0; c < 16; ++c) {
                    float pre = a0[c];
#pragma unroll
                    for (int f = 0; f < NF; ++f)
                        pre += g[f] * bj[f * MA * HID + c];
                    acc[c] += pre / (1.0f + __expf(-pre)); // SiLU
                }
            }
        }
#pragma unroll
        for (int c = 0; c < 16; ++c) sM[i * HID + cb + c] = acc[c];
    }
    __syncthreads();

    // ---- Stage C: upd = tanh([m|h] @ Wu + bu); h_new = h + 0.1*upd (WMMA) --
    // 8 i-tiles x 2 c-tiles, K=64 -> 2 chained wmma per tile.
    for (int t = wv; t < 16; t += 8) {
        const int jt   = t >> 1;
        const int ct   = t & 1;
        const int row0 = jt * 16 + m16;
        const int col  = ct * 16 + m16;

        v16h a0v, a1v, b0v, b1v;
        const float* xr0 = sM + row0 * HID;   // K 0..31  = m
        const float* xr1 = sH + row0 * HID;   // K 32..63 = h
#pragma unroll
        for (int u = 0; u < 8; ++u) {
            a0v[u]     = (_Float16)xr0[8 * hi + u];
            a0v[u + 8] = (_Float16)xr0[16 + 8 * hi + u];
            a1v[u]     = (_Float16)xr1[8 * hi + u];
            a1v[u + 8] = (_Float16)xr1[16 + 8 * hi + u];
        }
#pragma unroll
        for (int u = 0; u < 8; ++u) {
            b0v[u]     = (_Float16)Wu[(8 * hi + u) * HID + col];
            b0v[u + 8] = (_Float16)Wu[(16 + 8 * hi + u) * HID + col];
            b1v[u]     = (_Float16)Wu[(32 + 8 * hi + u) * HID + col];
            b1v[u + 8] = (_Float16)Wu[(48 + 8 * hi + u) * HID + col];
        }

        v8f acc = {};
        acc = __builtin_amdgcn_wmma_f32_16x16x32_f16(false, a0v, false, b0v, (short)0, acc, false, false);
        acc = __builtin_amdgcn_wmma_f32_16x16x32_f16(false, a1v, false, b1v, (short)0, acc, false, false);

        const float bcol = bu[col];
#pragma unroll
        for (int u = 0; u < 8; ++u) {
            const int rr = jt * 16 + 8 * hi + u;
            const float upd = tanhf(acc[u] + bcol);
            outH[(n * MA + rr) * HID + col] = sH[rr * HID + col] + UPDR * upd;
        }
    }
}

extern "C" void kernel_launch(void* const* d_in, const int* in_sizes, int n_in,
                              void* d_out, int out_size, void* d_ws, size_t ws_size,
                              hipStream_t stream) {
    (void)in_sizes; (void)n_in; (void)out_size; (void)d_ws; (void)ws_size;
    const int*   z    = (const int*)  d_in[0];
    const float* r    = (const float*)d_in[1];
    const float* h    = (const float*)d_in[2];
    const float* dist = (const float*)d_in[3];
    const float* wid  = (const float*)d_in[4];
    const float* Wc   = (const float*)d_in[5];
    const float* bc   = (const float*)d_in[6];
    const float* Wu   = (const float*)d_in[7];
    const float* bu   = (const float*)d_in[8];

    const size_t smem = (size_t)SMEM_FLOATS * sizeof(float);  // 313344 B (CDNA5 WGP has 320KB LDS)
    hipLaunchKernelGGL(MessagePassing_25786983645318_kernel,
                       dim3(NB), dim3(256), smem, stream,
                       z, r, h, dist, wid, Wc, bc, Wu, bu, (float*)d_out);
}
